// AttentionReadout_75376676045109
// MI455X (gfx1250) — compile-verified, matrix operand order checked by reading, not run on hardware
//
#include <hip/hip_runtime.h>
#include <hip/hip_bf16.h>
#include <math.h>

typedef __attribute__((ext_vector_type(16))) __bf16 v16bf;
typedef __attribute__((ext_vector_type(8)))  float  v8f;

#define IN_C  128
#define HID   64
#define NG    1024
#define OUT_C 128

// --- fast tanh: native V_TANH_F32 on gfx1250 ------------------------------
__device__ __forceinline__ float fast_tanh(float v) {
#if __has_builtin(__builtin_amdgcn_tanhf)
    return __builtin_amdgcn_tanhf(v);
#elif __has_builtin(__builtin_amdgcn_tanh_f32)
    return __builtin_amdgcn_tanh_f32(v);
#else
    float r;
    // TRANS op: 1 independent op required before result use -> pad with v_nop
    asm("v_tanh_f32 %0, %1\n\tv_nop" : "=v"(r) : "v"(v));
    return r;
#endif
}

// --- fast exp: single v_exp_f32 (scores are bounded, no range reduction) --
__device__ __forceinline__ float fast_exp(float v) {
#if __has_builtin(__builtin_amdgcn_exp2f)
    return __builtin_amdgcn_exp2f(v * 1.44269504088896340736f);
#else
    return __expf(v);
#endif
}

// ---------------------------------------------------------------------------
// Kernel 0: zero denom[NG] + pooled[NG*IN_C] (contiguous in workspace)
// ---------------------------------------------------------------------------
__global__ void zero_kernel(float* __restrict__ p, int n) {
    int i = blockIdx.x * blockDim.x + threadIdx.x;
    if (i < n) p[i] = 0.0f;
}

// ---------------------------------------------------------------------------
// Kernel 1: per-node attention score via bf16 WMMA MLP.
//   H = tanh(x @ W1 + b1)  [16x64 per wave-tile, 16 WMMAs]
//   s = H @ W2 + b2        [per-lane fold + 16-lane butterfly reduce]
//   e = exp(s); e_out[node] = e; denom[batch[node]] += e (atomic)
// ---------------------------------------------------------------------------
__global__ __launch_bounds__(256) void mlp_score_kernel(
    const float* __restrict__ x, const int* __restrict__ batch,
    const float* __restrict__ W1, const float* __restrict__ b1,
    const float* __restrict__ W2, const float* __restrict__ b2,
    float* __restrict__ e_out, float* __restrict__ denom, int N)
{
    const int lane  = threadIdx.x & 31;
    const int wave  = threadIdx.x >> 5;
    const int col   = lane & 15;            // B/C column held by this lane
    const int khalf = (lane >> 4) << 3;     // 0 (lanes 0-15) or 8 (lanes 16-31)

    // --- Preload W1 as 16 register-resident bf16 B-fragments (32x16 each) ---
    v16bf B[4][4];
#pragma unroll
    for (int nt = 0; nt < 4; ++nt)
#pragma unroll
        for (int ks = 0; ks < 4; ++ks)
#pragma unroll
            for (int j = 0; j < 16; ++j) {
                int kk = 32 * ks + ((j < 8) ? (khalf + j) : (16 + khalf + (j - 8)));
                B[nt][ks][j] = (__bf16)W1[kk * HID + nt * 16 + col];
            }

    float w2c[4], b1c[4];
#pragma unroll
    for (int nt = 0; nt < 4; ++nt) {
        w2c[nt] = W2[nt * 16 + col];
        b1c[nt] = b1[nt * 16 + col];
    }
    const float bias2 = b2[0];

    const int numTiles   = (N + 15) >> 4;
    const int waveGlobal = blockIdx.x * 8 + wave;
    const int waveStride = gridDim.x * 8;

    for (int tile = waveGlobal; tile < numTiles; tile += waveStride) {
        const int  base  = tile << 4;
        const int  row   = base + (lane & 15);   // A-matrix: lane -> row M
        const bool rowOk = row < N;

        // --- Prefetch the next grid-stride tile (no LOADcnt cost) ---------
        {
            const int nextTile = tile + waveStride;
            if (nextTile < numTiles) {
                int nrow = (nextTile << 4) + (lane & 15);
                if (nrow < N) {
                    // lanes 0-15 cover lines @ cols 0,64; lanes 16-31 @ 32,96
                    int off = (lane >> 4) << 5;
                    __builtin_prefetch(x + (size_t)nrow * IN_C + off, 0, 0);
                    __builtin_prefetch(x + (size_t)nrow * IN_C + 64 + off, 0, 0);
                }
            }
        }

        // --- Build 4 A-fragments (x rows, f32 -> bf16), float4 loads ------
        v16bf A[4];
        const float4* xr = (const float4*)(x + (size_t)(rowOk ? row : 0) * IN_C);
#pragma unroll
        for (int ks = 0; ks < 4; ++ks) {
            float4 f0 = xr[(32 * ks + khalf) >> 2];
            float4 f1 = xr[((32 * ks + khalf) >> 2) + 1];
            float4 f2 = xr[(32 * ks + 16 + khalf) >> 2];
            float4 f3 = xr[((32 * ks + 16 + khalf) >> 2) + 1];
            if (!rowOk) { f0 = f1 = f2 = f3 = make_float4(0.f, 0.f, 0.f, 0.f); }
            A[ks][0]  = (__bf16)f0.x; A[ks][1]  = (__bf16)f0.y;
            A[ks][2]  = (__bf16)f0.z; A[ks][3]  = (__bf16)f0.w;
            A[ks][4]  = (__bf16)f1.x; A[ks][5]  = (__bf16)f1.y;
            A[ks][6]  = (__bf16)f1.z; A[ks][7]  = (__bf16)f1.w;
            A[ks][8]  = (__bf16)f2.x; A[ks][9]  = (__bf16)f2.y;
            A[ks][10] = (__bf16)f2.z; A[ks][11] = (__bf16)f2.w;
            A[ks][12] = (__bf16)f3.x; A[ks][13] = (__bf16)f3.y;
            A[ks][14] = (__bf16)f3.z; A[ks][15] = (__bf16)f3.w;
        }

        // --- 4 n-tiles x 4 k-steps of WMMA; fold tanh(H)+bias into score ---
        float p[8] = {0.f, 0.f, 0.f, 0.f, 0.f, 0.f, 0.f, 0.f};
#pragma unroll
        for (int nt = 0; nt < 4; ++nt) {
            v8f c = {};
#pragma unroll
            for (int ks = 0; ks < 4; ++ks) {
                c = __builtin_amdgcn_wmma_f32_16x16x32_bf16(
                        false, A[ks], false, B[nt][ks], (short)0, c, false, false);
            }
            const float w2v = w2c[nt];
            const float b1v = b1c[nt];
#pragma unroll
            for (int r = 0; r < 8; ++r)
                p[r] += fast_tanh(c[r] + b1v) * w2v;
        }

        // --- Reduce over the 16 lanes of each half (columns) -> row scores ---
#pragma unroll
        for (int r = 0; r < 8; ++r) {
            float v = p[r];
            v += __shfl_xor(v, 1, 32);
            v += __shfl_xor(v, 2, 32);
            v += __shfl_xor(v, 4, 32);
            v += __shfl_xor(v, 8, 32);
            // C-layout: VGPR r holds row M = r (lanes 0-15) / r+8 (lanes 16-31)
            if ((lane & 15) == r) {
                int node = base + r + khalf;
                if (node < N) {
                    float ev = fast_exp(v + bias2);
                    e_out[node] = ev;
                    atomicAdd(&denom[batch[node]], ev);
                }
            }
        }
    }
}

// ---------------------------------------------------------------------------
// Kernel 2: unnormalized attention pooling: pooled[g][c] += e[i] * x[i][c].
// Each wave owns 64 sorted nodes; lane accumulates 4 channels (float4 loads),
// flush on graph change (wave-uniform branch) via global f32 atomics.
// ---------------------------------------------------------------------------
#define NODES_PER_WAVE 64
__global__ __launch_bounds__(128) void pool_kernel(
    const float* __restrict__ x, const int* __restrict__ batch,
    const float* __restrict__ e, float* __restrict__ pooled, int N)
{
    const int lane  = threadIdx.x & 31;
    const int wave  = threadIdx.x >> 5;
    const int chunk = blockIdx.x * 4 + wave;
    const int start = chunk * NODES_PER_WAVE;
    const int end   = min(start + NODES_PER_WAVE, N);
    if (start >= end) return;

    const int c4 = lane << 2;               // this lane's 4 channels
    int    curG = batch[start];
    float4 acc  = make_float4(0.f, 0.f, 0.f, 0.f);

#pragma unroll 4
    for (int i = start; i < end; ++i) {
        int g = batch[i];
        if (g != curG) {                    // wave-uniform
            float* dst = &pooled[curG * IN_C + c4];
            atomicAdd(dst + 0, acc.x); atomicAdd(dst + 1, acc.y);
            atomicAdd(dst + 2, acc.z); atomicAdd(dst + 3, acc.w);
            acc = make_float4(0.f, 0.f, 0.f, 0.f);
            curG = g;
        }
        float  ev = e[i];
        float4 xv = *(const float4*)(x + (size_t)i * IN_C + c4);
        acc.x += xv.x * ev; acc.y += xv.y * ev;
        acc.z += xv.z * ev; acc.w += xv.w * ev;
    }
    float* dst = &pooled[curG * IN_C + c4];
    atomicAdd(dst + 0, acc.x); atomicAdd(dst + 1, acc.y);
    atomicAdd(dst + 2, acc.z); atomicAdd(dst + 3, acc.w);
}

// ---------------------------------------------------------------------------
// Kernel 3: out[g] = (pooled[g] / denom[g]) @ Wt + bt via bf16 WMMA.
// 1024x128 @ 128x128 -> 64 m-tiles x 8 n-tiles = 512 wave-tiles.
// ---------------------------------------------------------------------------
__global__ __launch_bounds__(256) void out_gemm_kernel(
    const float* __restrict__ pooled, const float* __restrict__ denom,
    const float* __restrict__ Wt, const float* __restrict__ bt,
    float* __restrict__ out)
{
    const int lane  = threadIdx.x & 31;
    const int wave  = threadIdx.x >> 5;
    const int col   = lane & 15;
    const int khalf = (lane >> 4) << 3;

    const int tileId = blockIdx.x * 8 + wave;   // 0..511
    const int mt = tileId >> 3;                  // graph-row tile (0..63)
    const int nt = tileId & 7;                   // output-col tile (0..7)

    const int   grow = mt * 16 + (lane & 15);    // A row = graph id
    const float invd = 1.0f / denom[grow];       // normalize during A build

    v16bf A[4], B[4];
#pragma unroll
    for (int ks = 0; ks < 4; ++ks) {
#pragma unroll
        for (int j = 0; j < 16; ++j) {
            int kk = 32 * ks + ((j < 8) ? (khalf + j) : (16 + khalf + (j - 8)));
            A[ks][j] = (__bf16)(pooled[grow * IN_C + kk] * invd);
            B[ks][j] = (__bf16)Wt[kk * OUT_C + nt * 16 + col];
        }
    }

    v8f c = {};
#pragma unroll
    for (int ks = 0; ks < 4; ++ks)
        c = __builtin_amdgcn_wmma_f32_16x16x32_bf16(
                false, A[ks], false, B[ks], (short)0, c, false, false);

    const float btv = bt[nt * 16 + col];
#pragma unroll
    for (int r = 0; r < 8; ++r) {
        int orow = mt * 16 + r + khalf;
        out[orow * OUT_C + nt * 16 + col] = c[r] + btv;
    }
}

// ---------------------------------------------------------------------------
extern "C" void kernel_launch(void* const* d_in, const int* in_sizes, int n_in,
                              void* d_out, int out_size, void* d_ws, size_t ws_size,
                              hipStream_t stream) {
    const float* x     = (const float*)d_in[0];
    const int*   batch = (const int*)  d_in[1];
    const float* W1    = (const float*)d_in[2];
    const float* b1    = (const float*)d_in[3];
    const float* W2    = (const float*)d_in[4];
    const float* b2    = (const float*)d_in[5];
    const float* Wt    = (const float*)d_in[6];
    const float* bt    = (const float*)d_in[7];
    float*       out   = (float*)d_out;

    const int N = in_sizes[0] / IN_C;

    // Workspace layout: e[N] | denom[NG] | pooled[NG*IN_C]  (denom+pooled contiguous)
    float* e      = (float*)d_ws;
    float* denom  = e + N;
    float* pooled = denom + NG;

    const int zn = NG + NG * IN_C;
    zero_kernel<<<(zn + 255) / 256, 256, 0, stream>>>(denom, zn);

    mlp_score_kernel<<<512, 256, 0, stream>>>(x, batch, W1, b1, W2, b2, e, denom, N);

    const int nChunks = (N + NODES_PER_WAVE - 1) / NODES_PER_WAVE;
    pool_kernel<<<(nChunks + 3) / 4, 128, 0, stream>>>(x, batch, e, pooled, N);

    out_gemm_kernel<<<64, 256, 0, stream>>>(pooled, denom, Wt, bt, out);
}
